// HMPLayer_2826088481279
// MI455X (gfx1250) — matrix operand order repaired and checked by reading, not compiled, not run on hardware
//
#include <hip/hip_runtime.h>
#include <hip/hip_bf16.h>

// ---------------- problem constants (match reference) ----------------
#define NNODES   4096
#define FDIM     128
#define NEDGE    131072
#define MNODES   512
#define KTP      1152      // 9 * 128
#define HIDD     64
#define NBASIS   8

typedef __attribute__((ext_vector_type(16))) __bf16 v16bf;
typedef __attribute__((ext_vector_type(8)))  float  v8f;

// f32 -> bf16 round-to-nearest-even
__device__ inline unsigned short f2bf(float f) {
    unsigned u = __float_as_uint(f);
    unsigned r = u + 0x7FFFu + ((u >> 16) & 1u);
    return (unsigned short)(r >> 16);
}

// spherical harmonics (l<=2) + Bessel radial * p=6 cutoff, MACE-style
__device__ inline void geom_feats(float vx, float vy, float vz,
                                  float* __restrict__ sh, float* __restrict__ rad) {
    float ln = sqrtf(vx*vx + vy*vy + vz*vz);
    float nn = fmaxf(ln, 1e-9f);
    float x = vx/nn, y = vy/nn, z = vz/nn;
    const float s3 = 1.7320508f, s5 = 2.2360680f, s15 = 3.8729833f;
    sh[0] = 1.f;       sh[1] = s3*x;      sh[2] = s3*y;   sh[3] = s3*z;
    sh[4] = s15*x*y;   sh[5] = s15*y*z;   sh[6] = 0.5f*s5*(3.f*z*z - 1.f);
    sh[7] = s15*x*z;   sh[8] = 0.5f*s15*(x*x - y*y);
    float rs = fmaxf(ln, 1e-9f);
    float u  = fminf(fmaxf(ln * 0.2f, 0.f), 1.f);   // ln / R_MAX
    float u2 = u*u, u3 = u2*u, u6 = u3*u3, u7 = u6*u, u8 = u7*u;
    float cut = 1.f - 28.f*u6 + 48.f*u7 - 21.f*u8;
    const float c = 0.63245553f;                     // sqrt(2/R_MAX)
    float arg = 3.14159265358979f * rs * 0.2f;       // pi*rs/R_MAX
    #pragma unroll
    for (int n = 0; n < NBASIS; ++n)
        rad[n] = c * sinf((float)(n+1) * arg) / rs * cut;
}

// ---------------------------------------------------------------------
// K1b: pack Wtp (1152x128 f32 row-major) into WMMA B-fragment order.
// Fragment (kt,nt): lane l holds 16 bf16: B[kt*32 + (j>>3)*16 + (l>>4)*8 + (j&7), nt*16 + (l&15)]
// ---------------------------------------------------------------------
__global__ __launch_bounds__(256)
void pack_wtp_kernel(const float* __restrict__ Wtp, unsigned short* __restrict__ Bp) {
    int idx = blockIdx.x * 256 + threadIdx.x;
    if (idx >= KTP * FDIM) return;
    int j    = idx & 15;
    int lane = (idx >> 4) & 31;
    int nt   = (idx >> 9) & 7;
    int kt   = idx >> 12;
    int k    = kt*32 + (j >> 3)*16 + (lane >> 4)*8 + (j & 7);
    int col  = nt*16 + (lane & 15);
    Bp[idx] = f2bf(Wtp[k * FDIM + col]);
}

// ---------------------------------------------------------------------
// K1: per-edge: radial MLP gate, xw = h[src]*w, scatter sh⊗xw into Gn[dst]
// one block (128 threads) per edge
// ---------------------------------------------------------------------
__global__ __launch_bounds__(128)
void edge_conv_kernel(const float* __restrict__ h, const float* __restrict__ pos,
                      const int* __restrict__ ei,
                      const float* __restrict__ W1, const float* __restrict__ b1,
                      const float* __restrict__ W2, const float* __restrict__ b2,
                      float* __restrict__ Gn, float* __restrict__ cntv) {
    int e = blockIdx.x;
    int t = threadIdx.x;
    int src = ei[e];
    int dst = ei[NEDGE + e];
    float vx = pos[src*3+0] - pos[dst*3+0];
    float vy = pos[src*3+1] - pos[dst*3+1];
    float vz = pos[src*3+2] - pos[dst*3+2];
    float sh[9], rad[NBASIS];
    geom_feats(vx, vy, vz, sh, rad);

    __shared__ float hid[HIDD];
    if (t < HIDD) {
        float a = b1[t];
        #pragma unroll
        for (int n = 0; n < NBASIS; ++n) a += rad[n] * W1[n*HIDD + t];
        hid[t] = fmaxf(a, 0.f);
    }
    __syncthreads();
    float w = b2[t];
    for (int hh = 0; hh < HIDD; ++hh) w += hid[hh] * W2[hh*FDIM + t];
    float xw = h[(size_t)src*FDIM + t] * w;

    float* g = Gn + (size_t)dst*KTP + t;
    #pragma unroll
    for (int s = 0; s < 9; ++s) atomicAdd(g + s*FDIM, sh[s] * xw);
    if (t == 0) atomicAdd(cntv + dst, 1.0f);
}

// ---------------------------------------------------------------------
// K2/K10: bf16 WMMA GEMM: out = (A(rows x 1152) @ Bp) / max(cnt,1) + base
// one wave per 16-row strip x all 8 column tiles; 8 waves / block
// A-fragment layout per ISA §7.12.2 (16-bit A 16x32, wave32)
// ---------------------------------------------------------------------
__global__ __launch_bounds__(256)
void tp_gemm_kernel(const float* __restrict__ A, const unsigned short* __restrict__ Bp,
                    const float* __restrict__ cntv, const float* __restrict__ base,
                    float* __restrict__ out, int rows) {
    const int lane  = threadIdx.x & 31;
    const int wave  = threadIdx.x >> 5;
    const int strip = blockIdx.x * 8 + wave;
    if (strip * 16 >= rows) return;
    const int row0 = strip * 16;
    const int half = lane >> 4;
    const float* Arow = A + (size_t)(row0 + (lane & 15)) * KTP;

    v8f acc[8];
    #pragma unroll
    for (int nt = 0; nt < 8; ++nt)
        acc[nt] = (v8f){0.f,0.f,0.f,0.f,0.f,0.f,0.f,0.f};

    for (int kt = 0; kt < KTP/32; ++kt) {
        const float* p0 = Arow + kt*32 + half*8;
        const float* p1 = p0 + 16;
        __builtin_prefetch(p0 + 32, 0, 3);  // next K chunk -> global_prefetch
        union { v16bf v; unsigned short s[16]; } af;
        #pragma unroll
        for (int j = 0; j < 8; ++j) { af.s[j] = f2bf(p0[j]); af.s[8+j] = f2bf(p1[j]); }
        #pragma unroll
        for (int nt = 0; nt < 8; ++nt) {
            const v16bf b = *(const v16bf*)(const void*)(Bp + (((size_t)(kt*8 + nt)*32 + lane) << 4));
            acc[nt] = __builtin_amdgcn_wmma_f32_16x16x32_bf16(
                false, af.v, false, b, (short)0, acc[nt], false, false);
        }
    }
    // C/D layout: col = lane&15, row_in_tile = (lane>>4)*8 + r
    #pragma unroll
    for (int nt = 0; nt < 8; ++nt) {
        #pragma unroll
        for (int r = 0; r < 8; ++r) {
            int row = row0 + half*8 + r;
            int col = nt*16 + (lane & 15);
            float inv = 1.0f / fmaxf(cntv[row], 1.0f);
            size_t o = (size_t)row * FDIM + col;
            out[o] = acc[nt][r] * inv + base[o];
        }
    }
}

// ---------------------------------------------------------------------
// K3: selection MLP -> sigmoid score per node (block of 64 per node)
// ---------------------------------------------------------------------
__global__ __launch_bounds__(64)
void sel_kernel(const float* __restrict__ hl,
                const float* __restrict__ sW1, const float* __restrict__ sb1,
                const float* __restrict__ sW2, const float* __restrict__ sb2,
                float* __restrict__ mscore, float* __restrict__ out_m) {
    int n = blockIdx.x, t = threadIdx.x;
    const float* x = hl + (size_t)n * FDIM;
    float a = sb1[t];
    for (int f = 0; f < FDIM; ++f) a += x[f] * sW1[f*HIDD + t];
    a = fmaxf(a, 0.f) * sW2[t];
    __shared__ float red[HIDD];
    red[t] = a; __syncthreads();
    for (int s = 32; s > 0; s >>= 1) { if (t < s) red[t] += red[t+s]; __syncthreads(); }
    if (t == 0) {
        float z  = red[0] + sb2[0];
        float mm = 1.f / (1.f + expf(-z));
        mscore[n] = mm; out_m[n] = mm;
    }
}

// ---------------------------------------------------------------------
// K4: exact top-M via in-LDS bitonic sort (score desc, idx asc on ties),
// then sort chosen indices ascending; build rank[]. single 1024-thr block
// ---------------------------------------------------------------------
__global__ __launch_bounds__(1024)
void topk_kernel(const float* __restrict__ mscore,
                 int* __restrict__ master, int* __restrict__ rank) {
    __shared__ unsigned long long key[NNODES];
    __shared__ unsigned idxs[MNODES];
    int t = threadIdx.x;
    for (int i = t; i < NNODES; i += 1024) {
        unsigned u = __float_as_uint(mscore[i]);
        u = (u & 0x80000000u) ? ~u : (u | 0x80000000u);     // order-preserving map
        key[i] = ((unsigned long long)(~u) << 32) | (unsigned)i; // asc key = score desc, idx asc
    }
    __syncthreads();
    for (int k = 2; k <= NNODES; k <<= 1) {
        for (int j = k >> 1; j > 0; j >>= 1) {
            for (int i = t; i < NNODES/2; i += 1024) {
                int e = ((i & ~(j-1)) << 1) | (i & (j-1));
                int p = e ^ j;
                bool up = ((e & k) == 0);
                unsigned long long a = key[e], b = key[p];
                if ((a > b) == up) { key[e] = b; key[p] = a; }
            }
            __syncthreads();
        }
    }
    if (t < MNODES) idxs[t] = (unsigned)(key[t] & 0xFFFFFFFFu);
    __syncthreads();
    for (int k = 2; k <= MNODES; k <<= 1) {
        for (int j = k >> 1; j > 0; j >>= 1) {
            if (t < MNODES/2) {
                int i = t;
                int e = ((i & ~(j-1)) << 1) | (i & (j-1));
                int p = e ^ j;
                bool up = ((e & k) == 0);
                unsigned a = idxs[e], b = idxs[p];
                if ((a > b) == up) { idxs[e] = b; idxs[p] = a; }
            }
            __syncthreads();
        }
    }
    for (int i = t; i < NNODES; i += 1024) rank[i] = -1;
    __syncthreads();
    if (t < MNODES) { int n = (int)idxs[t]; master[t] = n; rank[n] = t; }
}

// K5: induced adjacency among masters (presence; races write same 1.0)
__global__ __launch_bounds__(256)
void adj_kernel(const int* __restrict__ ei, const int* __restrict__ rank,
                float* __restrict__ adj) {
    int e = blockIdx.x * 256 + threadIdx.x;
    if (e >= NEDGE) return;
    int r0 = rank[ei[e]], r1 = rank[ei[NEDGE + e]];
    if (r0 >= 0 && r1 >= 0) adj[(size_t)r0 * MNODES + r1] = 1.0f;
}

// K6: gather h_m/pos_m, compute q,k rows (block of 64 per master)
__global__ __launch_bounds__(64)
void qk_kernel(const float* __restrict__ hl, const float* __restrict__ pos,
               const int* __restrict__ master,
               const float* __restrict__ Wq, const float* __restrict__ Wk,
               float* __restrict__ hm, float* __restrict__ posm,
               float* __restrict__ qb, float* __restrict__ kb) {
    int j = blockIdx.x, t = threadIdx.x;
    int n = master[j];
    __shared__ float x[FDIM];
    x[t]      = hl[(size_t)n*FDIM + t];
    x[t + 64] = hl[(size_t)n*FDIM + t + 64];
    hm[(size_t)j*FDIM + t]      = x[t];
    hm[(size_t)j*FDIM + t + 64] = x[t + 64];
    if (t < 3) posm[j*3 + t] = pos[n*3 + t];
    __syncthreads();
    float q = 0.f, kk = 0.f;
    for (int f = 0; f < FDIM; ++f) { q += x[f]*Wq[f*HIDD+t]; kk += x[f]*Wk[f*HIDD+t]; }
    qb[(size_t)j*HIDD + t] = q;
    kb[(size_t)j*HIDD + t] = kk;
}

// K7: softmax row + threshold at 0.5*rowmax (== exp(s-max) >= 0.5) -> A_virtual
__global__ __launch_bounds__(256)
void attn_kernel(const float* __restrict__ qb, const float* __restrict__ kb,
                 float* __restrict__ Av) {
    int i = blockIdx.x, t = threadIdx.x;
    __shared__ float qr[HIDD];
    __shared__ float sc[MNODES];
    __shared__ float red[256];
    if (t < HIDD) qr[t] = qb[(size_t)i*HIDD + t];
    __syncthreads();
    for (int j = t; j < MNODES; j += 256) {
        const float* kr = kb + (size_t)j*HIDD;
        float s = 0.f;
        for (int f = 0; f < HIDD; ++f) s += qr[f]*kr[f];
        sc[j] = s * 0.125f;   // 1/sqrt(64)
    }
    __syncthreads();
    red[t] = fmaxf(sc[t], sc[t + 256]); __syncthreads();
    for (int s = 128; s > 0; s >>= 1) { if (t < s) red[t] = fmaxf(red[t], red[t+s]); __syncthreads(); }
    float mx = red[0];
    __syncthreads();
    float e0 = expf(sc[t] - mx), e1 = expf(sc[t + 256] - mx);
    red[t] = e0 + e1; __syncthreads();
    for (int s = 128; s > 0; s >>= 1) { if (t < s) red[t] += red[t+s]; __syncthreads(); }
    float inv = 1.f / red[0];
    Av[(size_t)i*MNODES + t]       = (e0 >= 0.5f) ? e0*inv : 0.f;
    Av[(size_t)i*MNODES + t + 256] = (e1 >= 0.5f) ? e1*inv : 0.f;
}

// K8: emask = adj | sym(Av off-diag); deg[j] = sum_i emask[i,j]
__global__ __launch_bounds__(256)
void emask_kernel(const float* __restrict__ adj, const float* __restrict__ Av,
                  float* __restrict__ emask, float* __restrict__ deg) {
    int j = blockIdx.x, t = threadIdx.x;
    float s = 0.f;
    for (int i = t; i < MNODES; i += 256) {
        bool mv = (i != j) && ((Av[(size_t)i*MNODES + j] > 0.f) || (Av[(size_t)j*MNODES + i] > 0.f));
        bool e  = (adj[(size_t)i*MNODES + j] > 0.f) || mv;
        float ev = e ? 1.f : 0.f;
        emask[(size_t)i*MNODES + j] = ev;
        s += ev;
    }
    __shared__ float red[256];
    red[t] = s; __syncthreads();
    for (int k = 128; k > 0; k >>= 1) { if (t < k) red[t] += red[t+k]; __syncthreads(); }
    if (t == 0) deg[j] = red[0];
}

// K9: G[j] = sum_i emask[i,j] * sh(i,j) ⊗ (h_m[i]*w2(i,j)); block per j,
// emask test is block-uniform -> skips inactive pairs (emask is very sparse)
__global__ __launch_bounds__(128)
void masterG_kernel(const float* __restrict__ hm, const float* __restrict__ posm,
                    const float* __restrict__ emask,
                    const float* __restrict__ W1, const float* __restrict__ b1,
                    const float* __restrict__ W2, const float* __restrict__ b2,
                    float* __restrict__ G) {
    int j = blockIdx.x, t = threadIdx.x;
    float acc[9];
    #pragma unroll
    for (int s = 0; s < 9; ++s) acc[s] = 0.f;
    float pjx = posm[j*3+0], pjy = posm[j*3+1], pjz = posm[j*3+2];
    __shared__ float hid[HIDD];
    for (int i = 0; i < MNODES; ++i) {
        if (emask[(size_t)i*MNODES + j] == 0.f) continue;   // uniform across block
        float sh[9], rad[NBASIS];
        geom_feats(posm[i*3+0]-pjx, posm[i*3+1]-pjy, posm[i*3+2]-pjz, sh, rad);
        if (t < HIDD) {
            float a = b1[t];
            #pragma unroll
            for (int n = 0; n < NBASIS; ++n) a += rad[n]*W1[n*HIDD + t];
            hid[t] = fmaxf(a, 0.f);
        }
        __syncthreads();
        float w = b2[t];
        for (int hh = 0; hh < HIDD; ++hh) w += hid[hh]*W2[hh*FDIM + t];
        float xw = hm[(size_t)i*FDIM + t] * w;
        #pragma unroll
        for (int s = 0; s < 9; ++s) acc[s] += sh[s]*xw;
        __syncthreads();
    }
    #pragma unroll
    for (int s = 0; s < 9; ++s) G[(size_t)j*KTP + s*FDIM + t] = acc[s];
}

// K11: gated fuse -> h_final
__global__ __launch_bounds__(256)
void fuse_kernel(const float* __restrict__ hl, const float* __restrict__ hhier,
                 const int* __restrict__ rank, const float* __restrict__ mscore,
                 float* __restrict__ out_h) {
    int idx = blockIdx.x * 256 + threadIdx.x;
    if (idx >= NNODES*FDIM) return;
    int n = idx >> 7, f = idx & 127;
    int r = rank[n];
    float he = (r >= 0) ? hhier[(size_t)r*FDIM + f] : 0.f;
    float mm = mscore[n];
    out_h[idx] = (1.f - mm)*hl[idx] + mm*he;
}

__global__ __launch_bounds__(256)
void poscopy_kernel(const float* __restrict__ pos, float* __restrict__ out_pos) {
    int idx = blockIdx.x * 256 + threadIdx.x;
    if (idx < NNODES*3) out_pos[idx] = pos[idx];
}

// ---------------------------------------------------------------------
extern "C" void kernel_launch(void* const* d_in, const int* in_sizes, int n_in,
                              void* d_out, int out_size, void* d_ws, size_t ws_size,
                              hipStream_t stream) {
    (void)in_sizes; (void)n_in; (void)out_size; (void)ws_size;
    const float* h     = (const float*)d_in[0];
    const float* pos   = (const float*)d_in[1];
    const int*   ei    = (const int*)  d_in[2];
    const float* W1    = (const float*)d_in[3];
    const float* b1    = (const float*)d_in[4];
    const float* W2    = (const float*)d_in[5];
    const float* b2    = (const float*)d_in[6];
    const float* Wtp   = (const float*)d_in[7];
    const float* sW1   = (const float*)d_in[8];
    const float* sb1   = (const float*)d_in[9];
    const float* sW2   = (const float*)d_in[10];
    const float* sb2   = (const float*)d_in[11];
    const float* Wq    = (const float*)d_in[12];
    const float* Wk    = (const float*)d_in[13];
    float* out = (float*)d_out;

    // output regions
    float* out_h   = out;
    float* out_pos = out + (size_t)NNODES*FDIM;
    float* out_Av  = out_pos + (size_t)NNODES*3;
    float* out_m   = out_Av + (size_t)MNODES*MNODES;

    // workspace layout (256B aligned regions)
    char* ws = (char*)d_ws;
    size_t o = 0;
    auto alloc = [&](size_t bytes) { char* p = ws + o; o += (bytes + 255) & ~(size_t)255; return p; };
    float*          Gn     = (float*)         alloc((size_t)NNODES*KTP*4);
    float*          cntv   = (float*)         alloc((size_t)NNODES*4);
    float*          hlocal = (float*)         alloc((size_t)NNODES*FDIM*4);
    unsigned short* Bp     = (unsigned short*)alloc((size_t)KTP*FDIM*2);
    float*          mscore = (float*)         alloc((size_t)NNODES*4);
    int*            master = (int*)           alloc((size_t)MNODES*4);
    int*            rank   = (int*)           alloc((size_t)NNODES*4);
    float*          adj    = (float*)         alloc((size_t)MNODES*MNODES*4);
    float*          hm     = (float*)         alloc((size_t)MNODES*FDIM*4);
    float*          posm   = (float*)         alloc((size_t)MNODES*3*4);
    float*          qb     = (float*)         alloc((size_t)MNODES*HIDD*4);
    float*          kb     = (float*)         alloc((size_t)MNODES*HIDD*4);
    float*          emask  = (float*)         alloc((size_t)MNODES*MNODES*4);
    float*          deg    = (float*)         alloc((size_t)MNODES*4);
    float*          G      = (float*)         alloc((size_t)MNODES*KTP*4);
    float*          hhier  = (float*)         alloc((size_t)MNODES*FDIM*4);

    // zero accumulators (capturable memset nodes)
    hipMemsetAsync(Gn,   0, (size_t)NNODES*KTP*4,    stream);
    hipMemsetAsync(cntv, 0, (size_t)NNODES*4,        stream);
    hipMemsetAsync(adj,  0, (size_t)MNODES*MNODES*4, stream);

    // stage 0: pack Wtp into WMMA B-fragment order (bf16)
    pack_wtp_kernel<<<(KTP*FDIM)/256, 256, 0, stream>>>(Wtp, Bp);

    // stage 1: edge conv -> Gn, cnt (scatter, L2 resident)
    edge_conv_kernel<<<NEDGE, 128, 0, stream>>>(h, pos, ei, W1, b1, W2, b2, Gn, cntv);

    // stage 2: h_local = (Gn @ Wtp)/max(cnt,1) + h   (bf16 WMMA)
    tp_gemm_kernel<<<(NNODES/16 + 7)/8, 256, 0, stream>>>(Gn, Bp, cntv, h, hlocal, NNODES);

    // stage 3: selection scores
    sel_kernel<<<NNODES, 64, 0, stream>>>(hlocal, sW1, sb1, sW2, sb2, mscore, out_m);

    // stage 4: exact top-M + rank
    topk_kernel<<<1, 1024, 0, stream>>>(mscore, master, rank);

    // stage 5: induced adjacency
    adj_kernel<<<NEDGE/256, 256, 0, stream>>>(ei, rank, adj);

    // stage 6: gather masters + q/k projections
    qk_kernel<<<MNODES, 64, 0, stream>>>(hlocal, pos, master, Wq, Wk, hm, posm, qb, kb);

    // stage 7: attention + threshold -> A_virtual (written straight to output)
    attn_kernel<<<MNODES, 256, 0, stream>>>(qb, kb, out_Av);

    // stage 8: emask + degree
    emask_kernel<<<MNODES, 256, 0, stream>>>(adj, out_Av, emask, deg);

    // stage 9: masked accumulation G[j] = sum_i emask * sh ⊗ (h_m * w2)
    masterG_kernel<<<MNODES, 128, 0, stream>>>(hm, posm, emask, W1, b1, W2, b2, G);

    // stage 10: h_hier = (G @ Wtp)/max(deg,1) + h_m   (bf16 WMMA)
    tp_gemm_kernel<<<(MNODES/16 + 7)/8, 256, 0, stream>>>(G, Bp, deg, hm, hhier, MNODES);

    // stage 11: gated fuse + pos passthrough
    fuse_kernel<<<(NNODES*FDIM)/256, 256, 0, stream>>>(hlocal, hhier, rank, mscore, out_h);
    poscopy_kernel<<<(NNODES*3 + 255)/256, 256, 0, stream>>>(pos, out_pos);
}
